// RecurrentHeteroGraphNN_63608465654040
// MI455X (gfx1250) — matrix operand order repaired and checked by reading, not compiled, not run on hardware
//
#include <hip/hip_runtime.h>
#include <hip/hip_bf16.h>
#include <math.h>

#define NUSER 100000
#define NITEM 100000
#define NEDGE 200000
#define DIM   128

typedef __attribute__((ext_vector_type(16))) __bf16 v16bf;
typedef __attribute__((ext_vector_type(8)))  float  v8f;

// ---------------------------------------------------------------------------
// WMMA helpers (CDNA5 v_wmma_f32_16x16x32_bf16, wave32)
// ---------------------------------------------------------------------------
__device__ __forceinline__ v8f wmma_bf16(v16bf a, v16bf b, v8f c) {
  return __builtin_amdgcn_wmma_f32_16x16x32_bf16(
      /*neg_a=*/false, a, /*neg_b=*/false, b,
      /*c_mod=*/(short)0, c, /*reuse_a=*/false, /*reuse_b=*/false);
}

// A-fragment: 16x32 bf16 tile from a row-major f32 matrix (row stride DIM),
// rows scaled by `s`. Layout per ISA 7.12.2 (16-bit A 16x32):
//   lanes 0-15 : M = lane,   v0..3 -> K pairs {0..7},  v4..7 -> K pairs {16..23}
//   lanes 16-31: M = lane-16, same VGPRs hold K pairs {8..15} / {24..31}
__device__ __forceinline__ v16bf load_a_frag(const float* __restrict__ base,
                                             int m0, int k0, float s, int lane) {
  const int half = lane >> 4;
  const int M    = m0 + (lane & 15);
  const float* row = base + (size_t)M * DIM + k0;
  v16bf a;
#pragma unroll
  for (int j = 0; j < 8; ++j) {
    const int kk = ((j < 4) ? (2 * j) : (16 + 2 * (j - 4))) + 8 * half;
    const float2 v = *(const float2*)(row + kk);
    a[2 * j]     = (__bf16)(v.x * s);
    a[2 * j + 1] = (__bf16)(v.y * s);
  }
  return a;
}

// B-fragment: 32x16 bf16 tile from a bf16 matrix stored as B[k][n] with row
// width ldn. Lane l holds column N = n0+(l&15); element s holds K = 8*half + s
// (s<8) or 16 + 8*half + (s-8) (s>=8), matching the A-fragment K order.
__device__ __forceinline__ v16bf load_b_frag(const __bf16* __restrict__ B,
                                             int ldn, int k0, int n0, int lane) {
  const int half = lane >> 4;
  const __bf16* col = B + (size_t)k0 * ldn + n0 + (lane & 15);
  v16bf b;
#pragma unroll
  for (int s = 0; s < 16; ++s) {
    const int kk = ((s < 8) ? s : (s + 8)) + 8 * half;
    b[s] = col[(size_t)kk * ldn];
  }
  return b;
}

// ---------------------------------------------------------------------------
// Utility kernels
// ---------------------------------------------------------------------------
__global__ void zero_f32(float* __restrict__ p, long long n) {
  for (long long i = (long long)blockIdx.x * blockDim.x + threadIdx.x; i < n;
       i += (long long)gridDim.x * blockDim.x)
    p[i] = 0.0f;
}

__global__ void deg_count(const int* __restrict__ idx, float* __restrict__ deg, int E) {
  for (long long i = (long long)blockIdx.x * blockDim.x + threadIdx.x; i < E;
       i += (long long)gridDim.x * blockDim.x)
    atomicAdd(deg + idx[i], 1.0f);
}

__global__ void finalize_rs(float* __restrict__ d, long long n) {
  for (long long i = (long long)blockIdx.x * blockDim.x + threadIdx.x; i < n;
       i += (long long)gridDim.x * blockDim.x)
    d[i] = rsqrtf(fmaxf(d[i], 1.0f));
}

// Transpose (rows x cols) f32 -> bf16 stored as out[c][r] (row width = rows)
__global__ void transpose_bf16(const float* __restrict__ w, __bf16* __restrict__ out,
                               int rows, int cols) {
  const long long n = (long long)rows * cols;
  for (long long i = (long long)blockIdx.x * blockDim.x + threadIdx.x; i < n;
       i += (long long)gridDim.x * blockDim.x) {
    const int r = (int)(i / cols), c = (int)(i % cols);
    out[(size_t)c * rows + r] = (__bf16)w[(size_t)r * cols + c];
  }
}

// Edge scatter: agg[dst] += x[src] * rs_o[src]. One lane handles 4 floats.
__global__ void scatter_agg(const float* __restrict__ x, const float* __restrict__ rs_o,
                            const int* __restrict__ src, const int* __restrict__ dst,
                            float* __restrict__ agg, int E) {
  const long long total = (long long)E * 32;
  for (long long i = (long long)blockIdx.x * blockDim.x + threadIdx.x; i < total;
       i += (long long)gridDim.x * blockDim.x) {
    const int e    = (int)(i >> 5);
    const int lane = (int)(i & 31);
    const int s = src[e], d = dst[e];
    const float sc = rs_o[s];
    const float4 v = ((const float4*)(x + (size_t)s * DIM))[lane];
    float* o = agg + (size_t)d * DIM + lane * 4;
    atomicAdd(o + 0, v.x * sc);
    atomicAdd(o + 1, v.y * sc);
    atomicAdd(o + 2, v.z * sc);
    atomicAdd(o + 3, v.w * sc);
  }
}

// ---------------------------------------------------------------------------
// Fused dual-gconv GEMM: out = act( (aggA*rsA)@WA + bA + (aggB*rsB)@WB + bB )
// One wave per 16-row tile x full 128 columns. Weights staged to LDS as bf16.
// ---------------------------------------------------------------------------
template <int RELU>
__global__ void __launch_bounds__(256)
gconv_gemm_pair(const float* __restrict__ aggA, const float* __restrict__ rsA,
                const float* __restrict__ WA,  const float* __restrict__ bA,
                const float* __restrict__ aggB, const float* __restrict__ rsB,
                const float* __restrict__ WB,  const float* __restrict__ bB,
                float* __restrict__ out, int ntiles) {
  extern __shared__ __bf16 ldsW[];  // 2 * 128*128 bf16 = 64 KB
  for (int i = threadIdx.x; i < DIM * DIM; i += blockDim.x) {
    ldsW[i]             = (__bf16)WA[i];
    ldsW[DIM * DIM + i] = (__bf16)WB[i];
  }
  __syncthreads();

  const int lane = threadIdx.x & 31;
  const int wave = threadIdx.x >> 5;
  const int tile = blockIdx.x * (blockDim.x >> 5) + wave;
  if (tile >= ntiles) return;  // wave-uniform: EXEC stays all-ones for WMMA

  const int m0   = tile * 16;
  const int nc   = lane & 15;
  const int half = lane >> 4;

  v8f acc[8];
#pragma unroll
  for (int t = 0; t < 8; ++t) {
    const float bv = bA[t * 16 + nc] + bB[t * 16 + nc];
#pragma unroll
    for (int r = 0; r < 8; ++r) acc[t][r] = bv;
  }

#pragma unroll
  for (int sel = 0; sel < 2; ++sel) {
    const float*  agg = sel ? aggB : aggA;
    const float*  rs  = sel ? rsB  : rsA;
    const __bf16* Wl  = ldsW + sel * (DIM * DIM);
    const float s = rs[m0 + nc];  // per-row rsqrt(deg_i) folded into A
#pragma unroll
    for (int k0 = 0; k0 < DIM; k0 += 32) {
      const v16bf a = load_a_frag(agg, m0, k0, s, lane);
#pragma unroll
      for (int t = 0; t < 8; ++t) {
        const v16bf b = load_b_frag(Wl, DIM, k0, t * 16, lane);
        acc[t] = wmma_bf16(a, b, acc[t]);
      }
    }
  }

#pragma unroll
  for (int t = 0; t < 8; ++t) {
#pragma unroll
    for (int r = 0; r < 8; ++r) {
      const int M = m0 + r + 8 * half;  // C/D layout: VGPR r -> M = r + 8*half
      float v = acc[t][r];
      if (RELU) v = fmaxf(v, 0.0f);
      out[(size_t)M * DIM + t * 16 + nc] = v;
    }
  }
}

// ---------------------------------------------------------------------------
// Fused GRU step: one wave per 16x16 output tile; 6 gate GEMMs accumulated in
// registers, gates applied in-register, single store of h'.
// ---------------------------------------------------------------------------
__global__ void __launch_bounds__(256)
gru_gemm(const float* __restrict__ x, const float* __restrict__ h0,
         const __bf16* __restrict__ wihT, const __bf16* __restrict__ whhT,
         const float* __restrict__ bih, const float* __restrict__ bhh,
         float* __restrict__ hout, int ntiles) {
  const int lane = threadIdx.x & 31;
  const int wave = threadIdx.x >> 5;
  const int w    = blockIdx.x * (blockDim.x >> 5) + wave;
  const int tile = w >> 3;
  const int t    = w & 7;
  if (tile >= ntiles) return;

  const int m0   = tile * 16;
  const int n0   = t * 16;
  const int nc   = lane & 15;
  const int half = lane >> 4;

  v8f air, aiz, ain, ahr, ahz, ahn;
#pragma unroll
  for (int r = 0; r < 8; ++r) {
    air[r] = bih[n0 + nc];
    aiz[r] = bih[DIM + n0 + nc];
    ain[r] = bih[2 * DIM + n0 + nc];
    ahr[r] = bhh[n0 + nc];
    ahz[r] = bhh[DIM + n0 + nc];
    ahn[r] = bhh[2 * DIM + n0 + nc];
  }

#pragma unroll
  for (int k0 = 0; k0 < DIM; k0 += 32) {
    const v16bf ax = load_a_frag(x,  m0, k0, 1.0f, lane);
    const v16bf ah = load_a_frag(h0, m0, k0, 1.0f, lane);
    air = wmma_bf16(ax, load_b_frag(wihT, 3 * DIM, k0, n0,           lane), air);
    aiz = wmma_bf16(ax, load_b_frag(wihT, 3 * DIM, k0, DIM + n0,     lane), aiz);
    ain = wmma_bf16(ax, load_b_frag(wihT, 3 * DIM, k0, 2 * DIM + n0, lane), ain);
    ahr = wmma_bf16(ah, load_b_frag(whhT, 3 * DIM, k0, n0,           lane), ahr);
    ahz = wmma_bf16(ah, load_b_frag(whhT, 3 * DIM, k0, DIM + n0,     lane), ahz);
    ahn = wmma_bf16(ah, load_b_frag(whhT, 3 * DIM, k0, 2 * DIM + n0, lane), ahn);
  }

#pragma unroll
  for (int r = 0; r < 8; ++r) {
    const int M = m0 + r + 8 * half;
    const float rg = 1.0f / (1.0f + __expf(-(air[r] + ahr[r])));
    const float zg = 1.0f / (1.0f + __expf(-(aiz[r] + ahz[r])));
    const float ng = tanhf(ain[r] + rg * ahn[r]);
    const float hp = h0[(size_t)M * DIM + n0 + nc];
    hout[(size_t)M * DIM + n0 + nc] = (1.0f - zg) * ng + zg * hp;
  }
}

// ---------------------------------------------------------------------------
// Host-side orchestration
// ---------------------------------------------------------------------------
extern "C" void kernel_launch(void* const* d_in, const int* in_sizes, int n_in,
                              void* d_out, int out_size, void* d_ws, size_t ws_size,
                              hipStream_t stream) {
  (void)in_sizes; (void)n_in; (void)out_size; (void)ws_size;

  const float* feat_user = (const float*)d_in[0];
  const float* feat_item = (const float*)d_in[1];
  const float* hidden0   = (const float*)d_in[2];

  // relations in setup_inputs order: 0=follows 1=buys 2=revbuys 3=similar
  const int *src[4], *dst[4];
  const float *W1[4], *b1[4], *W2[4], *b2[4];
  for (int r = 0; r < 4; ++r) {
    const int base = 3 + r * 6;
    src[r] = (const int*)d_in[base + 0];
    dst[r] = (const int*)d_in[base + 1];
    W1[r]  = (const float*)d_in[base + 2];
    b1[r]  = (const float*)d_in[base + 3];
    W2[r]  = (const float*)d_in[base + 4];
    b2[r]  = (const float*)d_in[base + 5];
  }
  const float* gru_wih = (const float*)d_in[27];
  const float* gru_whh = (const float*)d_in[28];
  const float* gru_bih = (const float*)d_in[29];
  const float* gru_bhh = (const float*)d_in[30];
  float* out = (float*)d_out;

  // Workspace layout (floats):
  //   [0, 800000)                      : rs arrays (4 rel x {o,i} x 100000)
  //   [800000, +4*100000*128)          : 4 agg buffers
  //   then 200000*128                  : h2 (GRU output)
  //   then 2 x 128*384 bf16            : wihT / whhT
  float* ws = (float*)d_ws;
  float* rs_o[4], * rs_i[4];
  for (int r = 0; r < 4; ++r) {
    rs_o[r] = ws + (size_t)(r * 2 + 0) * 100000;
    rs_i[r] = ws + (size_t)(r * 2 + 1) * 100000;
  }
  float* aggbase = ws + 800000;
  float* aggbuf[4];
  for (int r = 0; r < 4; ++r) aggbuf[r] = aggbase + (size_t)r * 100000 * DIM;
  float* h2 = aggbase + 4LL * 100000 * DIM;
  __bf16* wihT = (__bf16*)(h2 + (size_t)(NUSER + NITEM) * DIM);
  __bf16* whhT = wihT + 3 * DIM * DIM;

  const long long rs_agg_n = 800000LL + 4LL * 100000 * DIM;
  const long long agg_n    = 4LL * 100000 * DIM;

  // 0) zero rs + agg scratch
  zero_f32<<<8192, 256, 0, stream>>>(ws, rs_agg_n);

  // 1) degrees -> rsqrt(clip(deg,1))
  for (int r = 0; r < 4; ++r) {
    deg_count<<<512, 256, 0, stream>>>(src[r], rs_o[r], NEDGE);
    deg_count<<<512, 256, 0, stream>>>(dst[r], rs_i[r], NEDGE);
  }
  finalize_rs<<<1024, 256, 0, stream>>>(ws, 800000LL);

  // 2) pre-transpose GRU weights to bf16 B-layout [k][n]
  transpose_bf16<<<256, 256, 0, stream>>>(gru_wih, wihT, 3 * DIM, DIM);
  transpose_bf16<<<256, 256, 0, stream>>>(gru_whh, whhT, 3 * DIM, DIM);

  // 3) layer-1 edge aggregation
  scatter_agg<<<8192, 256, 0, stream>>>(feat_user, rs_o[0], src[0], dst[0], aggbuf[0], NEDGE); // follows
  scatter_agg<<<8192, 256, 0, stream>>>(feat_user, rs_o[1], src[1], dst[1], aggbuf[1], NEDGE); // buys
  scatter_agg<<<8192, 256, 0, stream>>>(feat_item, rs_o[2], src[2], dst[2], aggbuf[2], NEDGE); // revbuys
  scatter_agg<<<8192, 256, 0, stream>>>(feat_item, rs_o[3], src[3], dst[3], aggbuf[3], NEDGE); // similar

  // 4) layer-1 fused dual-gconv GEMM + relu  (writes GRU input into d_out)
  const int utiles = NUSER / 16, itiles = NITEM / 16;
  const int gblkU = (utiles + 7) / 8, gblkI = (itiles + 7) / 8;
  const size_t ldsBytes = 2u * DIM * DIM * sizeof(__bf16);  // 64 KB
  gconv_gemm_pair<1><<<gblkU, 256, ldsBytes, stream>>>(
      aggbuf[0], rs_i[0], W1[0], b1[0], aggbuf[2], rs_i[2], W1[2], b1[2],
      out, utiles);
  gconv_gemm_pair<1><<<gblkI, 256, ldsBytes, stream>>>(
      aggbuf[1], rs_i[1], W1[1], b1[1], aggbuf[3], rs_i[3], W1[3], b1[3],
      out + (size_t)NUSER * DIM, itiles);

  // 5) fused GRU step: x = d_out, h = hidden0 -> h2
  const int rtiles = (NUSER + NITEM) / 16;          // 12500
  gru_gemm<<<rtiles, 256, 0, stream>>>(out, hidden0, wihT, whhT,
                                       gru_bih, gru_bhh, h2, rtiles);

  // 6) re-zero agg, layer-2 aggregation from hu = h2[:NU], hi = h2[NU:]
  zero_f32<<<8192, 256, 0, stream>>>(aggbase, agg_n);
  const float* hu = h2;
  const float* hi = h2 + (size_t)NUSER * DIM;
  scatter_agg<<<8192, 256, 0, stream>>>(hu, rs_o[0], src[0], dst[0], aggbuf[0], NEDGE);
  scatter_agg<<<8192, 256, 0, stream>>>(hu, rs_o[1], src[1], dst[1], aggbuf[1], NEDGE);
  scatter_agg<<<8192, 256, 0, stream>>>(hi, rs_o[2], src[2], dst[2], aggbuf[2], NEDGE);
  scatter_agg<<<8192, 256, 0, stream>>>(hi, rs_o[3], src[3], dst[3], aggbuf[3], NEDGE);

  // 7) layer-2 fused dual-gconv GEMM (no relu) -> d_out
  gconv_gemm_pair<0><<<gblkU, 256, ldsBytes, stream>>>(
      aggbuf[0], rs_i[0], W2[0], b2[0], aggbuf[2], rs_i[2], W2[2], b2[2],
      out, utiles);
  gconv_gemm_pair<0><<<gblkI, 256, ldsBytes, stream>>>(
      aggbuf[1], rs_i[1], W2[1], b2[1], aggbuf[3], rs_i[3], W2[3], b2[3],
      out + (size_t)NUSER * DIM, itiles);
}
